// myBlock_58282706206861
// MI455X (gfx1250) — matrix operand order repaired and checked by reading, not compiled, not run on hardware
//
#include <hip/hip_runtime.h>
#include <hip/hip_bf16.h>

typedef _Float16 v16h __attribute__((ext_vector_type(16)));
typedef _Float16 v8h  __attribute__((ext_vector_type(8)));
typedef float    v8f  __attribute__((ext_vector_type(8)));
typedef unsigned int v4u __attribute__((ext_vector_type(4)));
typedef int      v8i  __attribute__((ext_vector_type(8)));
typedef int      v4i  __attribute__((ext_vector_type(4)));

#define WMMA16(a, b, c) __builtin_amdgcn_wmma_f32_16x16x32_f16( \
    false, (a), false, (b), (short)0, (c), false, false)

#ifndef __has_builtin
#define __has_builtin(x) 0
#endif
#if __has_builtin(__builtin_amdgcn_tensor_load_to_lds) && \
    __has_builtin(__builtin_amdgcn_s_wait_tensorcnt)
#define USE_TDM 1
#else
#define USE_TDM 0
#endif

// ---------------------------------------------------------------------------
// Problem constants (match reference)
// ---------------------------------------------------------------------------
#define BATCH   128
#define NTOK    197
#define CDIM    768
#define HDIM    3072
#define NHEAD   12
#define HD      64
#define MKEEP   178     // 1 + (196 - 19)
#define MPAD    192     // MKEEP padded to multiple of 16
#define TROWS   (BATCH * MKEEP)   // 22784 = 16 * 1424
#define THR     19

// ---------------------------------------------------------------------------
// WMMA operand loaders (ISA 7.12.2 wave32 layouts)
// A 16x32 f16: lane L holds row (L&15); halves 0..7 = K[b..b+7],
//              halves 8..15 = K[b+16..b+23], b = 8*(L>=16)
// B 32x16 f16: lane L holds col (L&15); same K order as A
// ---------------------------------------------------------------------------
__device__ __forceinline__ v16h load_a_tile(const _Float16* p0, int ld, int lane) {
  const _Float16* p = p0 + (size_t)(lane & 15) * ld + ((lane >> 4) << 3);
  v8h c0 = *(const v8h*)(p);
  v8h c1 = *(const v8h*)(p + 16);
  v16h a;
#pragma unroll
  for (int i = 0; i < 8; ++i) { a[i] = c0[i]; a[i + 8] = c1[i]; }
  return a;
}

__device__ __forceinline__ v16h load_b_tile(const _Float16* p0, int ld, int lane) {
  const _Float16* p = p0 + (size_t)(lane & 15) * ld + ((lane >> 4) << 3);
  v8h c0 = *(const v8h*)(p);
  v8h c1 = *(const v8h*)(p + 16);
  v16h b;
#pragma unroll
  for (int i = 0; i < 8; ++i) { b[i] = c0[i]; b[i + 8] = c1[i]; }
  return b;
}

__device__ __forceinline__ v8f vzero8() {
  v8f z = {0.f, 0.f, 0.f, 0.f, 0.f, 0.f, 0.f, 0.f};
  return z;
}

// ---------------------------------------------------------------------------
// TDM: async-load a 2D tile [tile_n rows x tile_k elems] of f16 (row stride
// row_stride elems) from global into LDS at lds_off.  ISA ch.8 D# layout:
//   group0: {count=1 | lds_addr | global_addr[56:0] | type=2}
//   group1: {data_size=2B | tensor dims (huge) | tile dims | dim0 stride}
// 6-arg builtin form (amdgpu-toolchain / clang-23).
// ---------------------------------------------------------------------------
#if USE_TDM
__device__ __forceinline__ void tdm_load_2d(const _Float16* gptr, unsigned lds_off,
                                            int tile_k, int tile_n, int row_stride) {
  unsigned long long ga = (unsigned long long)(const void*)gptr;
  v4u g0;
  g0[0] = 1u;                                      // count=1, user descriptor
  g0[1] = lds_off;                                 // LDS byte address
  g0[2] = (unsigned)(ga & 0xffffffffu);            // global_addr[31:0]
  g0[3] = (unsigned)((ga >> 32) & 0x01ffffffu)     // global_addr[56:32]
        | (2u << 30);                              // type = 2 ("image")
  const unsigned td0 = 0x7fffffffu, td1 = 0x7fffffffu;  // no OOB clipping
  v8i g1;
  g1[0] = (int)(1u << 16);                         // data_size = 1 -> 2 bytes
  g1[1] = (int)((td0 & 0xffffu) << 16);            // tensor_dim0[15:0]
  g1[2] = (int)(((td0 >> 16) & 0xffffu) | ((td1 & 0xffffu) << 16));
  g1[3] = (int)(((td1 >> 16) & 0xffffu) | ((unsigned)tile_k << 16)); // tile_dim0
  g1[4] = (int)((unsigned)tile_n & 0xffffu);       // tile_dim1 (tile_dim2 = 0)
  g1[5] = (int)(unsigned)row_stride;               // tensor_dim0_stride[31:0]
  g1[6] = 0;                                       // stride hi / dim1 stride
  g1[7] = 0;
  v4i z4 = {0, 0, 0, 0};
  v8i z8 = {0, 0, 0, 0, 0, 0, 0, 0};
  __builtin_amdgcn_tensor_load_to_lds(g0, g1, z4, z4, z8, 0);
}
#endif

// ---------------------------------------------------------------------------
// Block GEMM body: 8 waves, block tile 128(M) x 64(N), K in 64-wide chunks.
// Wave computes its own 16x64 D.  B chunk [64N x 64K] staged in LDS,
// double-buffered via TDM (fallback: direct global B loads).
// A: per-wave row-major f16 pointer (already offset to this wave's m0).
// Wn0: weights row-major [N][K], already offset to n0*K.
// ---------------------------------------------------------------------------
__device__ __forceinline__ void gemm_body(const _Float16* A, int lda,
                                          const _Float16* Wn0, int K,
                                          _Float16* lbuf, int lane, int wave,
                                          v8f acc[4]) {
#if USE_TDM
  const int NC = K >> 6;
  const unsigned lbase = (unsigned)(unsigned long long)(void*)lbuf;
  if (wave == 0) tdm_load_2d(Wn0, lbase, 64, 64, K);
  for (int c = 0; c < NC; ++c) {
    const int cur = c & 1;
    if (wave == 0) {
      if (c + 1 < NC) {
        tdm_load_2d(Wn0 + (c + 1) * 64, lbase + (unsigned)((cur ^ 1) * 64 * 64 * 2),
                    64, 64, K);
        __builtin_amdgcn_s_wait_tensorcnt(1);
      } else {
        __builtin_amdgcn_s_wait_tensorcnt(0);
      }
    }
    __syncthreads();                       // B chunk `cur` visible to all waves
    const _Float16* Bb = lbuf + cur * 64 * 64;
#pragma unroll
    for (int ks = 0; ks < 2; ++ks) {
      v16h a = load_a_tile(A + c * 64 + ks * 32, lda, lane);
#pragma unroll
      for (int j = 0; j < 4; ++j) {
        v16h b = load_b_tile(Bb + (size_t)(16 * j) * 64 + ks * 32, 64, lane);
        acc[j] = WMMA16(a, b, acc[j]);
      }
    }
    __syncthreads();                       // reads done before chunk is reused
  }
#else
  (void)lbuf; (void)wave;
  for (int k0 = 0; k0 < K; k0 += 32) {
    v16h a = load_a_tile(A + k0, lda, lane);
#pragma unroll
    for (int j = 0; j < 4; ++j) {
      v16h b = load_b_tile(Wn0 + (size_t)(16 * j) * K + k0, K, lane);
      acc[j] = WMMA16(a, b, acc[j]);
    }
  }
#endif
}

// ---------------------------------------------------------------------------
// Kernel: f32 -> f16 conversion (weights)
// ---------------------------------------------------------------------------
__global__ __launch_bounds__(256) void k_cvt(const float* __restrict__ s,
                                             _Float16* __restrict__ d, int n) {
  int i = blockIdx.x * 256 + threadIdx.x;
  if (i < n) d[i] = (_Float16)s[i];
}

// ---------------------------------------------------------------------------
// Kernel: token drop -> kept index table [BATCH][MKEEP]
// ---------------------------------------------------------------------------
__global__ __launch_bounds__(256) void k_drop(const float* __restrict__ rnd,
                                              int* __restrict__ kept) {
  __shared__ float r[NTOK - 1];
  __shared__ unsigned char keep[NTOK - 1];
  const int b = blockIdx.x, j = threadIdx.x;
  if (j < NTOK - 1) r[j] = rnd[(size_t)b * (NTOK - 1) + j];
  __syncthreads();
  if (j < NTOK - 1) {
    float rj = r[j];
    int rank = 0;
    for (int i = 0; i < NTOK - 1; ++i)
      rank += (r[i] < rj) || (r[i] == rj && i < j);
    keep[j] = (rank >= THR) ? 1 : 0;
  }
  __syncthreads();
  if (j < NTOK - 1 && keep[j]) {
    int pos = 0;
    for (int i = 0; i < j; ++i) pos += keep[i];
    kept[b * MKEEP + 1 + pos] = j + 1;
  }
  if (j == 0) kept[b * MKEEP] = 0;
}

// ---------------------------------------------------------------------------
// Kernel: (optional gather) + LayerNorm. One block per token row.
// ---------------------------------------------------------------------------
__global__ __launch_bounds__(256) void k_ln(const float* __restrict__ xin,
                                            const int* __restrict__ kept,
                                            const float* __restrict__ g,
                                            const float* __restrict__ bt,
                                            float* __restrict__ xout,
                                            _Float16* __restrict__ hout) {
  __shared__ float sA[256], sB[256];
  const int t = blockIdx.x, tid = threadIdx.x;
  const float* row;
  if (kept) {
    int b = t / MKEEP;
    row = xin + ((size_t)b * NTOK + kept[t]) * CDIM;
  } else {
    row = xin + (size_t)t * CDIM;
  }
  float v[3], lsum = 0.f, lsq = 0.f;
#pragma unroll
  for (int i = 0; i < 3; ++i) {
    v[i] = row[i * 256 + tid];
    lsum += v[i];
    lsq += v[i] * v[i];
  }
  sA[tid] = lsum; sB[tid] = lsq;
  __syncthreads();
  for (int s = 128; s > 0; s >>= 1) {
    if (tid < s) { sA[tid] += sA[tid + s]; sB[tid] += sB[tid + s]; }
    __syncthreads();
  }
  const float mean = sA[0] * (1.0f / CDIM);
  const float var  = sB[0] * (1.0f / CDIM) - mean * mean;
  const float rstd = rsqrtf(var + 1e-5f);
#pragma unroll
  for (int i = 0; i < 3; ++i) {
    int c = i * 256 + tid;
    float y = (v[i] - mean) * rstd * g[c] + bt[c];
    hout[(size_t)t * CDIM + c] = (_Float16)y;
    if (xout) xout[(size_t)t * CDIM + c] = v[i];
  }
}

// ---------------------------------------------------------------------------
// Kernel: QKV GEMM, scatter epilogue into q/k [b][h][MPAD][64] and
// transposed v [b][h][64][MPAD].  Q pre-scaled by 0.125.
// ---------------------------------------------------------------------------
__global__ __launch_bounds__(256) void k_qkv(const _Float16* __restrict__ ln1,
                                             const _Float16* __restrict__ wq,
                                             _Float16* __restrict__ qb,
                                             _Float16* __restrict__ kb,
                                             _Float16* __restrict__ vt) {
  __shared__ alignas(32) _Float16 lbuf[2 * 64 * 64];
  const int lane = threadIdx.x & 31, wave = threadIdx.x >> 5;
  const int m0 = (blockIdx.y * 8 + wave) * 16;
  const int n0 = blockIdx.x * 64;
  v8f acc[4];
#pragma unroll
  for (int j = 0; j < 4; ++j) acc[j] = vzero8();
  gemm_body(ln1 + (size_t)m0 * CDIM, CDIM, wq + (size_t)n0 * CDIM, CDIM,
            lbuf, lane, wave, acc);
  const int rb = (lane >> 4) << 3, nc = lane & 15;
#pragma unroll
  for (int j = 0; j < 4; ++j) {
#pragma unroll
    for (int r = 0; r < 8; ++r) {
      int m = m0 + r + rb;
      int n = n0 + 16 * j + nc;
      float val = acc[j][r];
      int bb = m / MKEEP, mm = m - bb * MKEEP;
      if (n < 768) {
        int h = n >> 6, d = n & 63;
        qb[(((size_t)(bb * NHEAD + h) * MPAD + mm) << 6) + d] = (_Float16)(val * 0.125f);
      } else if (n < 1536) {
        int o = n - 768, h = o >> 6, d = o & 63;
        kb[(((size_t)(bb * NHEAD + h) * MPAD + mm) << 6) + d] = (_Float16)val;
      } else {
        int o = n - 1536, h = o >> 6, d = o & 63;
        vt[((size_t)(bb * NHEAD + h) * HD + d) * MPAD + mm] = (_Float16)val;
      }
    }
  }
}

// ---------------------------------------------------------------------------
// Kernel: attention. One wave per 16-query tile; 192 padded keys, masked.
// ---------------------------------------------------------------------------
__global__ __launch_bounds__(128) void k_attn(const _Float16* __restrict__ qb,
                                              const _Float16* __restrict__ kb,
                                              const _Float16* __restrict__ vt,
                                              _Float16* __restrict__ attn) {
  __shared__ _Float16 lp[4 * 16 * 200];
  const int lane = threadIdx.x & 31, wave = threadIdx.x >> 5;
  const int b = blockIdx.z, h = blockIdx.y;
  const int m0 = (blockIdx.x * 4 + wave) * 16;
  const _Float16* qbh = qb + (size_t)(b * NHEAD + h) * MPAD * HD;
  const _Float16* kbh = kb + (size_t)(b * NHEAD + h) * MPAD * HD;
  const _Float16* vbh = vt + (size_t)(b * NHEAD + h) * HD * MPAD;
  const int rb = (lane >> 4) << 3, nc = lane & 15;

  v16h aq[2];
#pragma unroll
  for (int ds = 0; ds < 2; ++ds)
    aq[ds] = load_a_tile(qbh + (size_t)m0 * HD + ds * 32, HD, lane);

  v8f s[12];
#pragma unroll
  for (int kt = 0; kt < 12; ++kt) s[kt] = vzero8();
#pragma unroll
  for (int kt = 0; kt < 12; ++kt) {
#pragma unroll
    for (int ds = 0; ds < 2; ++ds) {
      v16h bt = load_b_tile(kbh + (size_t)(kt * 16) * HD + ds * 32, HD, lane);
      s[kt] = WMMA16(aq[ds], bt, s[kt]);
    }
  }
#pragma unroll
  for (int kt = 0; kt < 12; ++kt) {
    if (kt * 16 + nc >= MKEEP) {
#pragma unroll
      for (int r = 0; r < 8; ++r) s[kt][r] = -1e30f;
    }
  }
  float mx[8], sminv[8];
#pragma unroll
  for (int r = 0; r < 8; ++r) {
    float m = -1e30f;
#pragma unroll
    for (int kt = 0; kt < 12; ++kt) m = fmaxf(m, s[kt][r]);
    m = fmaxf(m, __shfl_xor(m, 1, 32));
    m = fmaxf(m, __shfl_xor(m, 2, 32));
    m = fmaxf(m, __shfl_xor(m, 4, 32));
    m = fmaxf(m, __shfl_xor(m, 8, 32));
    mx[r] = m;
  }
#pragma unroll
  for (int r = 0; r < 8; ++r) {
    float t = 0.f;
#pragma unroll
    for (int kt = 0; kt < 12; ++kt) {
      float e = __expf(s[kt][r] - mx[r]);
      s[kt][r] = e;
      t += e;
    }
    t += __shfl_xor(t, 1, 32);
    t += __shfl_xor(t, 2, 32);
    t += __shfl_xor(t, 4, 32);
    t += __shfl_xor(t, 8, 32);
    sminv[r] = 1.0f / (t + 1e-20f);
  }
  _Float16* myp = lp + wave * 16 * 200;
#pragma unroll
  for (int kt = 0; kt < 12; ++kt) {
#pragma unroll
    for (int r = 0; r < 8; ++r)
      myp[(r + rb) * 200 + kt * 16 + nc] = (_Float16)(s[kt][r] * sminv[r]);
  }
  __syncthreads();
  v8f o[4];
#pragma unroll
  for (int j = 0; j < 4; ++j) o[j] = vzero8();
  for (int ks = 0; ks < 6; ++ks) {
    v16h a = load_a_tile(myp + ks * 32, 200, lane);
#pragma unroll
    for (int j = 0; j < 4; ++j) {
      v16h bt = load_b_tile(vbh + (size_t)(j * 16) * MPAD + ks * 32, MPAD, lane);
      o[j] = WMMA16(a, bt, o[j]);
    }
  }
#pragma unroll
  for (int j = 0; j < 4; ++j) {
#pragma unroll
    for (int r = 0; r < 8; ++r) {
      int m = m0 + r + rb;
      if (m < MKEEP)
        attn[((size_t)(b * MKEEP + m)) * CDIM + h * HD + j * 16 + nc] = (_Float16)(o[j][r]);
    }
  }
}

// ---------------------------------------------------------------------------
// Kernel: proj GEMM + bias + residual -> x2 (f32)
// ---------------------------------------------------------------------------
__global__ __launch_bounds__(256) void k_proj(const _Float16* __restrict__ attn,
                                              const _Float16* __restrict__ wp,
                                              const float* __restrict__ bproj,
                                              const float* __restrict__ x1,
                                              float* __restrict__ x2) {
  __shared__ alignas(32) _Float16 lbuf[2 * 64 * 64];
  const int lane = threadIdx.x & 31, wave = threadIdx.x >> 5;
  const int m0 = (blockIdx.y * 8 + wave) * 16;
  const int n0 = blockIdx.x * 64;
  v8f acc[4];
#pragma unroll
  for (int j = 0; j < 4; ++j) acc[j] = vzero8();
  gemm_body(attn + (size_t)m0 * CDIM, CDIM, wp + (size_t)n0 * CDIM, CDIM,
            lbuf, lane, wave, acc);
  const int rb = (lane >> 4) << 3, nc = lane & 15;
#pragma unroll
  for (int j = 0; j < 4; ++j) {
#pragma unroll
    for (int r = 0; r < 8; ++r) {
      int m = m0 + r + rb, n = n0 + 16 * j + nc;
      size_t idx = (size_t)m * CDIM + n;
      x2[idx] = acc[j][r] + bproj[n] + x1[idx];
    }
  }
}

// ---------------------------------------------------------------------------
// Kernel: FC1 GEMM + bias + exact GELU -> f16
// ---------------------------------------------------------------------------
__global__ __launch_bounds__(256) void k_fc1(const _Float16* __restrict__ ln2,
                                             const _Float16* __restrict__ w1,
                                             const float* __restrict__ b1,
                                             _Float16* __restrict__ h1) {
  __shared__ alignas(32) _Float16 lbuf[2 * 64 * 64];
  const int lane = threadIdx.x & 31, wave = threadIdx.x >> 5;
  const int m0 = (blockIdx.y * 8 + wave) * 16;
  const int n0 = blockIdx.x * 64;
  v8f acc[4];
#pragma unroll
  for (int j = 0; j < 4; ++j) acc[j] = vzero8();
  gemm_body(ln2 + (size_t)m0 * CDIM, CDIM, w1 + (size_t)n0 * CDIM, CDIM,
            lbuf, lane, wave, acc);
  const int rb = (lane >> 4) << 3, nc = lane & 15;
#pragma unroll
  for (int j = 0; j < 4; ++j) {
#pragma unroll
    for (int r = 0; r < 8; ++r) {
      int m = m0 + r + rb, n = n0 + 16 * j + nc;
      float g = acc[j][r] + b1[n];
      float y = 0.5f * g * (1.0f + erff(g * 0.70710678118654752f));
      h1[(size_t)m * HDIM + n] = (_Float16)y;
    }
  }
}

// ---------------------------------------------------------------------------
// Kernel: FC2 GEMM + bias + residual -> out (f32)
// ---------------------------------------------------------------------------
__global__ __launch_bounds__(256) void k_fc2(const _Float16* __restrict__ h1,
                                             const _Float16* __restrict__ w2,
                                             const float* __restrict__ b2,
                                             const float* __restrict__ x2,
                                             float* __restrict__ out) {
  __shared__ alignas(32) _Float16 lbuf[2 * 64 * 64];
  const int lane = threadIdx.x & 31, wave = threadIdx.x >> 5;
  const int m0 = (blockIdx.y * 8 + wave) * 16;
  const int n0 = blockIdx.x * 64;
  v8f acc[4];
#pragma unroll
  for (int j = 0; j < 4; ++j) acc[j] = vzero8();
  gemm_body(h1 + (size_t)m0 * HDIM, HDIM, w2 + (size_t)n0 * HDIM, HDIM,
            lbuf, lane, wave, acc);
  const int rb = (lane >> 4) << 3, nc = lane & 15;
#pragma unroll
  for (int j = 0; j < 4; ++j) {
#pragma unroll
    for (int r = 0; r < 8; ++r) {
      int m = m0 + r + rb, n = n0 + 16 * j + nc;
      size_t idx = (size_t)m * CDIM + n;
      out[idx] = x2[idx] + acc[j][r] + b2[n];
    }
  }
}

// ---------------------------------------------------------------------------
// Host launch
// ---------------------------------------------------------------------------
extern "C" void kernel_launch(void* const* d_in, const int* in_sizes, int n_in,
                              void* d_out, int out_size, void* d_ws, size_t ws_size,
                              hipStream_t stream) {
  (void)in_sizes; (void)n_in; (void)out_size; (void)ws_size;
  const float* x      = (const float*)d_in[0];
  const float* rnd    = (const float*)d_in[1];
  const float* w_qkv  = (const float*)d_in[2];
  const float* w_proj = (const float*)d_in[3];
  const float* b_proj = (const float*)d_in[4];
  const float* g1     = (const float*)d_in[5];
  const float* b1     = (const float*)d_in[6];
  const float* g2     = (const float*)d_in[7];
  const float* b2     = (const float*)d_in[8];
  const float* w_fc1  = (const float*)d_in[9];
  const float* b_fc1  = (const float*)d_in[10];
  const float* w_fc2  = (const float*)d_in[11];
  const float* b_fc2  = (const float*)d_in[12];
  float* out = (float*)d_out;

  char* ws = (char*)d_ws;
  size_t off = 0;
  auto alloc = [&](size_t bytes) -> void* {
    void* p = ws + off;
    off += (bytes + 255) & ~(size_t)255;
    return p;
  };
  int*       kept = (int*)       alloc((size_t)BATCH * MKEEP * 4);
  _Float16*  wq16 = (_Float16*)  alloc((size_t)3 * CDIM * CDIM * 2);
  _Float16*  wp16 = (_Float16*)  alloc((size_t)CDIM * CDIM * 2);
  _Float16*  w116 = (_Float16*)  alloc((size_t)HDIM * CDIM * 2);
  _Float16*  w216 = (_Float16*)  alloc((size_t)CDIM * HDIM * 2);
  float*     x1   = (float*)     alloc((size_t)TROWS * CDIM * 4);
  _Float16*  ln1  = (_Float16*)  alloc((size_t)TROWS * CDIM * 2);
  _Float16*  qb   = (_Float16*)  alloc((size_t)BATCH * NHEAD * MPAD * HD * 2);
  _Float16*  kb   = (_Float16*)  alloc((size_t)BATCH * NHEAD * MPAD * HD * 2);
  _Float16*  vt   = (_Float16*)  alloc((size_t)BATCH * NHEAD * HD * MPAD * 2);
  _Float16*  attn = (_Float16*)  alloc((size_t)TROWS * CDIM * 2);
  float*     x2   = (float*)     alloc((size_t)TROWS * CDIM * 4);
  _Float16*  ln2  = (_Float16*)  alloc((size_t)TROWS * CDIM * 2);
  _Float16*  h1   = (_Float16*)  alloc((size_t)TROWS * HDIM * 2);

  {
    int n;
    n = 3 * CDIM * CDIM; k_cvt<<<(n + 255) / 256, 256, 0, stream>>>(w_qkv, wq16, n);
    n = CDIM * CDIM;     k_cvt<<<(n + 255) / 256, 256, 0, stream>>>(w_proj, wp16, n);
    n = HDIM * CDIM;     k_cvt<<<(n + 255) / 256, 256, 0, stream>>>(w_fc1, w116, n);
    n = CDIM * HDIM;     k_cvt<<<(n + 255) / 256, 256, 0, stream>>>(w_fc2, w216, n);
  }
  k_drop<<<BATCH, 256, 0, stream>>>(rnd, kept);
  k_ln<<<TROWS, 256, 0, stream>>>(x, kept, g1, b1, x1, ln1);
  k_qkv<<<dim3(3 * CDIM / 64, TROWS / 128), 256, 0, stream>>>(ln1, wq16, qb, kb, vt);
  k_attn<<<dim3(MPAD / 64, NHEAD, BATCH), 128, 0, stream>>>(qb, kb, vt, attn);
  k_proj<<<dim3(CDIM / 64, TROWS / 128), 256, 0, stream>>>(attn, wp16, b_proj, x1, x2);
  k_ln<<<TROWS, 256, 0, stream>>>(x2, nullptr, g2, b2, nullptr, ln2);
  k_fc1<<<dim3(HDIM / 64, TROWS / 128), 256, 0, stream>>>(ln2, w116, b_fc1, h1);
  k_fc2<<<dim3(CDIM / 64, TROWS / 128), 256, 0, stream>>>(h1, w216, b_fc2, x2, out);
}
